// EnhancedCryptoTransformer_24077586662055
// MI455X (gfx1250) — compile-verified, matrix operand order checked by reading, not run on hardware
//
#include <hip/hip_runtime.h>
#include <math.h>

// ---------------------------------------------------------------------------
// EnhancedCryptoTransformer forward for MI455X (gfx1250, wave32, WMMA).
//
// d_in order (setup_inputs dict order, layers dict flattened in insertion
// order):
//  0 x[8000,64] 1 w_in[64,512] 2 b_in[512] 3 pe[250,512] 4 pos_scale[250]
//  5 pos_bias[250] 6 tbias[250,250]
//  7 wq[8,512,512] 8 bq[8,512] 9 wk 10 bk 11 wv 12 bv 13 wo 14 bo
//  15 g1[8,512,256] 16 gb1[8,256] 17 g2[8,256,1] 18 gb2[8,1]
//  19 lna_s[8,512] 20 lna_b 21 n1s 22 n1b
//  23 f1[8,512,2048] 24 fb1[8,2048] 25 f2[8,2048,512] 26 fb2[8,512]
//  27 n2s 28 n2b 29 final_scale[512] 30 final_bias[512]
// ---------------------------------------------------------------------------

#define BB 32
#define SS 250
#define DD 512
#define HH 16
#define LL 8
#define DFF 2048
#define DK 32
#define MROWS (BB * SS)   // 8000
#define SPAD 256

typedef __bf16 bf16x16 __attribute__((ext_vector_type(16)));
typedef float  floatx8 __attribute__((ext_vector_type(8)));

struct U128 { unsigned int a, b, c, d; };
union BF16Frag { bf16x16 v; unsigned short u[16]; U128 q[2]; };

__device__ inline unsigned short f2bf(float f) {
  unsigned int u = __float_as_uint(f);
  unsigned int r = u + 0x7FFFu + ((u >> 16) & 1u);   // round-to-nearest-even
  return (unsigned short)(r >> 16);
}
__device__ inline float bf2f(unsigned short s) {
  return __uint_as_float(((unsigned int)s) << 16);
}

// Async global->LDS B128 copy (CDNA5 GLOBAL_LOAD_ASYNC_TO_LDS_B128, ASYNCcnt).
// ldsOff = byte offset into LDS (low 32 bits of the generic shared address).
__device__ inline void async_copy_b128(void* ldsDst, const void* gSrc) {
  unsigned int ldsOff = (unsigned int)(size_t)ldsDst;
  asm volatile("global_load_async_to_lds_b128 %0, %1, off"
               :: "v"(ldsOff), "v"(gSrc) : "memory");
}
__device__ inline void async_wait0() {
  asm volatile("s_wait_asynccnt 0" ::: "memory");
}

// ---------------------------------------------------------------------------
// Generic bf16 WMMA GEMM: C[M,N] = act(A[M,K](bf16) @ W[K,N](f32) + bias)
// Tile 64x128, 256 threads = 8 waves (2x4), wave -> 32x32 (2x2 wmma), BK=32.
// A tile staged via async-LDS DMA; W staged fp32->bf16 transposed [n][k].
// EPI: 0 = none, 1 = relu, 2 = exact gelu.
// ---------------------------------------------------------------------------
template <int EPI>
__global__ __launch_bounds__(256) void gemm_bf16_kernel(
    const unsigned short* __restrict__ A, const float* __restrict__ W,
    const float* __restrict__ bias, float* __restrict__ Cf,
    unsigned short* __restrict__ Cb, int M, int N, int K, int lda) {
  __shared__ __align__(16) unsigned short As[64 * 32];    // [m][k]
  __shared__ __align__(16) unsigned short Bs[128 * 32];   // [n][k] (transposed)

  const int tid  = threadIdx.x;
  const int lane = tid & 31;
  const int wid  = tid >> 5;
  const int wm   = (wid >> 2) * 32;
  const int wn   = (wid & 3) * 32;
  const int m0   = blockIdx.y * 64;
  const int n0   = blockIdx.x * 128;
  const int kb   = (lane >> 4) * 8;      // 0 or 8 (A/B frag K-chunk base)
  const int ln15 = lane & 15;
  const int hi8  = (lane >> 4) << 3;

  floatx8 acc[2][2] = {};
  const int ksteps = K >> 5;

  for (int kt = 0; kt < ksteps; ++kt) {
    const int k0 = kt << 5;
    // Stage A tile async: 64 rows x 32 bf16 = 256 x 16B chunks, one per thread.
    {
      const int r = tid >> 2, chunk = (tid & 3) * 8;
      async_copy_b128(As + r * 32 + chunk,
                      A + (size_t)(m0 + r) * lda + k0 + chunk);
    }
    // Stage W tile transposed (f32 -> bf16): Bs[n][k], 16 cols per thread.
    {
      const int kr = tid >> 3, c0 = (tid & 7) * 16;
      const float* src = W + (size_t)(k0 + kr) * N + n0 + c0;
#pragma unroll
      for (int i = 0; i < 16; ++i) Bs[(c0 + i) * 32 + kr] = f2bf(src[i]);
    }
    if (kt + 1 < ksteps)   // global_prefetch_b8 next W K-slab
      __builtin_prefetch(W + (size_t)(k0 + 32 + (tid >> 3)) * N + n0, 0, 1);
    async_wait0();
    __syncthreads();

    BF16Frag a[2], b[2];
#pragma unroll
    for (int mi = 0; mi < 2; ++mi) {
      const U128* p = (const U128*)(As + (wm + mi * 16 + ln15) * 32 + kb);
      a[mi].q[0] = p[0];   // K = kb .. kb+7
      a[mi].q[1] = p[2];   // K = kb+16 .. kb+23
    }
#pragma unroll
    for (int ni = 0; ni < 2; ++ni) {
      const U128* p = (const U128*)(Bs + (wn + ni * 16 + ln15) * 32 + kb);
      b[ni].q[0] = p[0];
      b[ni].q[1] = p[2];
    }
#pragma unroll
    for (int mi = 0; mi < 2; ++mi)
#pragma unroll
      for (int ni = 0; ni < 2; ++ni)
        acc[mi][ni] = __builtin_amdgcn_wmma_f32_16x16x32_bf16(
            false, a[mi].v, false, b[ni].v, (short)0, acc[mi][ni], false, false);
    __syncthreads();
  }

  // Epilogue: bias + activation, optional f32 / bf16 stores.
#pragma unroll
  for (int mi = 0; mi < 2; ++mi)
#pragma unroll
    for (int ni = 0; ni < 2; ++ni) {
#pragma unroll
      for (int r = 0; r < 8; ++r) {
        const int row = m0 + wm + mi * 16 + r + hi8;
        const int col = n0 + wn + ni * 16 + ln15;
        float v = acc[mi][ni][r] + bias[col];
        if (EPI == 1) v = fmaxf(v, 0.0f);
        if (EPI == 2) v = 0.5f * v * (1.0f + erff(v * 0.70710678118654752f));
        const size_t o = (size_t)row * N + col;
        if (Cf) Cf[o] = v;
        if (Cb) Cb[o] = f2bf(v);
      }
    }
}

// ---------------------------------------------------------------------------
// Fused attention per (b,h): scores = (Q K^T)/sqrt(DK) + tbias, softmax, P V.
// Block = 128 threads (4 waves); each wave owns one 16-row query tile;
// blockIdx.x in [0,4) -> 16 query tiles cover SPAD=256 rows.
// K rows staged via async-LDS DMA; V transposed; P relayout via wave scratch.
// ---------------------------------------------------------------------------
__global__ __launch_bounds__(128) void attention_kernel(
    const unsigned short* __restrict__ Q, const unsigned short* __restrict__ Km,
    const unsigned short* __restrict__ Vm, const float* __restrict__ tbias,
    unsigned short* __restrict__ O) {
  __shared__ __align__(16) unsigned short Ks[SPAD * 32];   // [s][k]
  __shared__ __align__(16) unsigned short Vt[32 * SPAD];   // [d][s]
  __shared__ __align__(16) unsigned short Pscr[4][16 * 32];

  const int bh   = blockIdx.y;          // 0..511
  const int b    = bh >> 4;
  const int h    = bh & 15;
  const int tid  = threadIdx.x;
  const int lane = tid & 31;
  const int wid  = tid >> 5;
  const int kb   = (lane >> 4) * 8;
  const int ln15 = lane & 15;
  const int hi8  = (lane >> 4) << 3;
  const float scale = 0.17677669529663687f;   // 1/sqrt(32)

  // Stage K (async DMA rows) and V (transposed), zero-fill pad rows 250..255.
  for (int s = tid; s < SPAD; s += 128) {
    if (s < SS) {
      const unsigned short* ksrc = Km + (size_t)(b * SS + s) * DD + h * 32;
#pragma unroll
      for (int c = 0; c < 4; ++c)
        async_copy_b128(Ks + s * 32 + c * 8, ksrc + c * 8);
      const unsigned short* vs = Vm + (size_t)(b * SS + s) * DD + h * 32;
#pragma unroll
      for (int d = 0; d < 32; ++d) Vt[d * SPAD + s] = vs[d];
    } else {
      const U128 z = {0u, 0u, 0u, 0u};
      U128* kd = (U128*)(Ks + s * 32);
      kd[0] = z; kd[1] = z; kd[2] = z; kd[3] = z;
#pragma unroll
      for (int d = 0; d < 32; ++d) Vt[d * SPAD + s] = 0;
    }
  }
  async_wait0();
  __syncthreads();

  const int qt    = blockIdx.x * 4 + wid;   // query tile 0..15
  const int qbase = qt * 16;

  // Q fragment (16x32 bf16 A-layout): two contiguous 16B chunks per lane.
  BF16Frag qa;
  {
    const int mrow = qbase + ln15;
    if (mrow < SS) {
      const U128* p = (const U128*)(Q + (size_t)(b * SS + mrow) * DD + h * 32 + kb);
      qa.q[0] = p[0];
      qa.q[1] = p[2];
    } else {
      const U128 z = {0u, 0u, 0u, 0u};
      qa.q[0] = z; qa.q[1] = z;
    }
  }

  // Scores: 16 key tiles of 16 columns each.
  floatx8 sc[16];
#pragma unroll
  for (int jt = 0; jt < 16; ++jt) {
    BF16Frag kf;
    const U128* p = (const U128*)(Ks + (jt * 16 + ln15) * 32 + kb);
    kf.q[0] = p[0];
    kf.q[1] = p[2];
    floatx8 z = {};
    sc[jt] = __builtin_amdgcn_wmma_f32_16x16x32_bf16(
        false, qa.v, false, kf.v, (short)0, z, false, false);
  }

  // Scale + additive time-decay bias; pad columns get -inf-ish.
#pragma unroll
  for (int jt = 0; jt < 16; ++jt) {
    const int col = jt * 16 + ln15;
#pragma unroll
    for (int r = 0; r < 8; ++r) {
      const int row = qbase + r + hi8;
      float v = -1.0e30f;
      if (row < SS && col < SS)
        v = sc[jt][r] * scale + tbias[(size_t)row * SS + col];
      sc[jt][r] = v;
    }
  }

  // Softmax per row: each row lives in one 16-lane half; reduce with xor 8..1.
#pragma unroll
  for (int r = 0; r < 8; ++r) {
    float m = -1.0e30f;
#pragma unroll
    for (int jt = 0; jt < 16; ++jt) m = fmaxf(m, sc[jt][r]);
    for (int msk = 8; msk; msk >>= 1) m = fmaxf(m, __shfl_xor(m, msk, 32));
    float s = 0.0f;
#pragma unroll
    for (int jt = 0; jt < 16; ++jt) {
      const float e = __expf(sc[jt][r] - m);
      sc[jt][r] = e;
      s += e;
    }
    for (int msk = 8; msk; msk >>= 1) s += __shfl_xor(s, msk, 32);
    const float inv = 1.0f / s;
#pragma unroll
    for (int jt = 0; jt < 16; ++jt) sc[jt][r] *= inv;
  }

  // P @ V in chunks of 32 keys; P relayout D-layout -> A-layout via LDS.
  floatx8 oacc[2] = {};
  unsigned short* myP = &Pscr[wid][0];
#pragma unroll
  for (int c = 0; c < 8; ++c) {
#pragma unroll
    for (int half = 0; half < 2; ++half) {
      const int jt = 2 * c + half;
#pragma unroll
      for (int r = 0; r < 8; ++r)
        myP[(r + hi8) * 32 + half * 16 + ln15] = f2bf(sc[jt][r]);
    }
    asm volatile("s_wait_dscnt 0" ::: "memory");   // wave-private RAW fence
    BF16Frag pa;
    {
      const U128* p = (const U128*)(myP + ln15 * 32 + kb);
      pa.q[0] = p[0];
      pa.q[1] = p[2];
    }
#pragma unroll
    for (int nh = 0; nh < 2; ++nh) {
      BF16Frag vf;   // B[k][n] = V[32c+k][nh*16+ln15] = Vt[n][32c+k], contiguous
      const U128* p = (const U128*)(Vt + (nh * 16 + ln15) * SPAD + 32 * c + kb);
      vf.q[0] = p[0];
      vf.q[1] = p[2];
      oacc[nh] = __builtin_amdgcn_wmma_f32_16x16x32_bf16(
          false, pa.v, false, vf.v, (short)0, oacc[nh], false, false);
    }
  }

  // Store O (bf16, [B*S, D] layout).
#pragma unroll
  for (int nh = 0; nh < 2; ++nh)
#pragma unroll
    for (int r = 0; r < 8; ++r) {
      const int row = qbase + r + hi8;
      if (row < SS) {
        const int col = h * 32 + nh * 16 + ln15;
        O[(size_t)(b * SS + row) * DD + col] = f2bf(oacc[nh][r]);
      }
    }
}

// ---------------------------------------------------------------------------
// Gate stage 2: g = sigmoid(relu_hidden[8000,256](bf16) @ g2[256] + gb2)
// One wave per row.
// ---------------------------------------------------------------------------
__global__ __launch_bounds__(256) void gate2_kernel(
    const unsigned short* __restrict__ gh, const float* __restrict__ w,
    const float* __restrict__ b, float* __restrict__ g, int M) {
  const int lane = threadIdx.x & 31;
  const int wid  = threadIdx.x >> 5;
  const int row  = blockIdx.x * 8 + wid;
  if (row >= M) return;
  const unsigned short* p = gh + (size_t)row * 256 + lane * 8;
  float acc = 0.0f;
#pragma unroll
  for (int i = 0; i < 8; ++i) acc += bf2f(p[i]) * w[lane * 8 + i];
  for (int msk = 16; msk; msk >>= 1) acc += __shfl_xor(acc, msk, 32);
  if (lane == 0) g[row] = 1.0f / (1.0f + __expf(-(acc + b[0])));
}

// ---------------------------------------------------------------------------
// Row reductions for LayerNorm (128 threads = 4 waves).
// ---------------------------------------------------------------------------
__device__ inline float block_sum(float v, float* red, int tid) {
  for (int msk = 16; msk; msk >>= 1) v += __shfl_xor(v, msk, 32);
  if ((tid & 31) == 0) red[tid >> 5] = v;
  __syncthreads();
  const float s = red[0] + red[1] + red[2] + red[3];
  __syncthreads();
  return s;
}

// attn_out = LN(o*g + h*(1-g)); h1 = LN(h + attn_out); h,hbf updated in place.
__global__ __launch_bounds__(128) void ln_blend_kernel(
    float* __restrict__ h, const float* __restrict__ o,
    const float* __restrict__ g, const float* __restrict__ as,
    const float* __restrict__ ab, const float* __restrict__ ns,
    const float* __restrict__ nb, unsigned short* __restrict__ hbf) {
  __shared__ float red[4];
  const int row = blockIdx.x, tid = threadIdx.x;
  const size_t base = (size_t)row * DD;
  const float gv = g[row];
  float hv[4], tv[4];
#pragma unroll
  for (int i = 0; i < 4; ++i) {
    const int c = tid + i * 128;
    hv[i] = h[base + c];
    tv[i] = o[base + c] * gv + hv[i] * (1.0f - gv);
  }
  float s = tv[0] + tv[1] + tv[2] + tv[3];
  float q = tv[0] * tv[0] + tv[1] * tv[1] + tv[2] * tv[2] + tv[3] * tv[3];
  const float mean = block_sum(s, red, tid) * (1.0f / DD);
  const float var  = block_sum(q, red, tid) * (1.0f / DD) - mean * mean;
  const float rstd = rsqrtf(var + 1e-5f);
  float uv[4];
#pragma unroll
  for (int i = 0; i < 4; ++i) {
    const int c = tid + i * 128;
    const float a = (tv[i] - mean) * rstd * as[c] + ab[c];
    uv[i] = hv[i] + a;
  }
  s = uv[0] + uv[1] + uv[2] + uv[3];
  q = uv[0] * uv[0] + uv[1] * uv[1] + uv[2] * uv[2] + uv[3] * uv[3];
  const float mean2 = block_sum(s, red, tid) * (1.0f / DD);
  const float var2  = block_sum(q, red, tid) * (1.0f / DD) - mean2 * mean2;
  const float rstd2 = rsqrtf(var2 + 1e-5f);
#pragma unroll
  for (int i = 0; i < 4; ++i) {
    const int c = tid + i * 128;
    const float r = (uv[i] - mean2) * rstd2 * ns[c] + nb[c];
    h[base + c]   = r;
    hbf[base + c] = f2bf(r);
  }
}

// out = LN(x + add?) with scale/bias; optional f32 / bf16 outputs.
__global__ __launch_bounds__(128) void ln_kernel(
    const float* __restrict__ x, const float* __restrict__ add,
    const float* __restrict__ sc, const float* __restrict__ bi,
    float* __restrict__ outf, unsigned short* __restrict__ outbf) {
  __shared__ float red[4];
  const int row = blockIdx.x, tid = threadIdx.x;
  const size_t base = (size_t)row * DD;
  float uv[4];
#pragma unroll
  for (int i = 0; i < 4; ++i) {
    const int c = tid + i * 128;
    uv[i] = x[base + c] + (add ? add[base + c] : 0.0f);
  }
  float s = uv[0] + uv[1] + uv[2] + uv[3];
  float q = uv[0] * uv[0] + uv[1] * uv[1] + uv[2] * uv[2] + uv[3] * uv[3];
  const float mean = block_sum(s, red, tid) * (1.0f / DD);
  const float var  = block_sum(q, red, tid) * (1.0f / DD) - mean * mean;
  const float rstd = rsqrtf(var + 1e-5f);
#pragma unroll
  for (int i = 0; i < 4; ++i) {
    const int c = tid + i * 128;
    const float r = (uv[i] - mean) * rstd * sc[c] + bi[c];
    if (outf)  outf[base + c]  = r;
    if (outbf) outbf[base + c] = f2bf(r);
  }
}

// h = (h + pe[s]) * pos_scale[s] + pos_bias[s]; also write bf16 shadow.
__global__ void pe_kernel(float* __restrict__ h, unsigned short* __restrict__ hbf,
                          const float* __restrict__ pe, const float* __restrict__ ps,
                          const float* __restrict__ pb) {
  const size_t idx = (size_t)blockIdx.x * blockDim.x + threadIdx.x;
  if (idx >= (size_t)MROWS * DD) return;
  const int row = (int)(idx / DD);
  const int s   = row % SS;
  const int c   = (int)(idx % DD);
  const float v = (h[idx] + pe[(size_t)s * DD + c]) * ps[s] + pb[s];
  h[idx]   = v;
  hbf[idx] = f2bf(v);
}

__global__ void f2bf_kernel(const float* __restrict__ x,
                            unsigned short* __restrict__ y, size_t n) {
  const size_t idx = (size_t)blockIdx.x * blockDim.x + threadIdx.x;
  if (idx < n) y[idx] = f2bf(x[idx]);
}

// ---------------------------------------------------------------------------
// Host side
// ---------------------------------------------------------------------------
static void launch_gemm(int epi, const unsigned short* A, const float* W,
                        const float* bias, float* Cf, unsigned short* Cb,
                        int M, int N, int K, int lda, hipStream_t st) {
  dim3 grid(N / 128, M / 64), blk(256);
  if (epi == 1)
    gemm_bf16_kernel<1><<<grid, blk, 0, st>>>(A, W, bias, Cf, Cb, M, N, K, lda);
  else if (epi == 2)
    gemm_bf16_kernel<2><<<grid, blk, 0, st>>>(A, W, bias, Cf, Cb, M, N, K, lda);
  else
    gemm_bf16_kernel<0><<<grid, blk, 0, st>>>(A, W, bias, Cf, Cb, M, N, K, lda);
}

extern "C" void kernel_launch(void* const* d_in, const int* in_sizes, int n_in,
                              void* d_out, int out_size, void* d_ws, size_t ws_size,
                              hipStream_t stream) {
  const float* x      = (const float*)d_in[0];
  const float* w_in   = (const float*)d_in[1];
  const float* b_in   = (const float*)d_in[2];
  const float* pe     = (const float*)d_in[3];
  const float* ps     = (const float*)d_in[4];
  const float* pb     = (const float*)d_in[5];
  const float* tbias  = (const float*)d_in[6];
  const float* wq     = (const float*)d_in[7];
  const float* bq     = (const float*)d_in[8];
  const float* wk     = (const float*)d_in[9];
  const float* bk     = (const float*)d_in[10];
  const float* wv     = (const float*)d_in[11];
  const float* bv     = (const float*)d_in[12];
  const float* wo     = (const float*)d_in[13];
  const float* bo     = (const float*)d_in[14];
  const float* g1     = (const float*)d_in[15];
  const float* gb1    = (const float*)d_in[16];
  const float* g2     = (const float*)d_in[17];
  const float* gb2    = (const float*)d_in[18];
  const float* lna_s  = (const float*)d_in[19];
  const float* lna_b  = (const float*)d_in[20];
  const float* n1s    = (const float*)d_in[21];
  const float* n1b    = (const float*)d_in[22];
  const float* f1     = (const float*)d_in[23];
  const float* fb1    = (const float*)d_in[24];
  const float* f2     = (const float*)d_in[25];
  const float* fb2    = (const float*)d_in[26];
  const float* n2s    = (const float*)d_in[27];
  const float* n2b    = (const float*)d_in[28];
  const float* fscale = (const float*)d_in[29];
  const float* fbias  = (const float*)d_in[30];

  // Workspace layout (bytes); ffmid overlays q/k/v/o, ff overlays attn_raw.
  char* ws = (char*)d_ws;
  const size_t OFF_H     = 0;                          // f32 [8000,512]
  const size_t OFF_HBF   = OFF_H     + (size_t)MROWS * DD * 4;
  const size_t OFF_Q     = OFF_HBF   + (size_t)MROWS * DD * 2;
  const size_t OFF_K     = OFF_Q     + (size_t)MROWS * DD * 2;
  const size_t OFF_V     = OFF_K     + (size_t)MROWS * DD * 2;
  const size_t OFF_O     = OFF_V     + (size_t)MROWS * DD * 2;
  const size_t OFF_ATTN  = OFF_O     + (size_t)MROWS * DD * 2;  // f32, also ff
  const size_t OFF_GHID  = OFF_ATTN  + (size_t)MROWS * DD * 4;  // bf16 [8000,256]
  const size_t OFF_G     = OFF_GHID  + (size_t)MROWS * 256 * 2; // f32 [8000]
  const size_t OFF_XBF   = OFF_G     + (size_t)MROWS * 4;
  const size_t OFF_FFMID = OFF_Q;                               // bf16 [8000,2048]

  float*          h     = (float*)(ws + OFF_H);
  unsigned short* hbf   = (unsigned short*)(ws + OFF_HBF);
  unsigned short* qbf   = (unsigned short*)(ws + OFF_Q);
  unsigned short* kbf   = (unsigned short*)(ws + OFF_K);
  unsigned short* vbf   = (unsigned short*)(ws + OFF_V);
  unsigned short* obf   = (unsigned short*)(ws + OFF_O);
  float*          attn  = (float*)(ws + OFF_ATTN);
  float*          ff    = (float*)(ws + OFF_ATTN);
  unsigned short* ghid  = (unsigned short*)(ws + OFF_GHID);
  float*          gvec  = (float*)(ws + OFF_G);
  unsigned short* xbf   = (unsigned short*)(ws + OFF_XBF);
  unsigned short* ffmid = (unsigned short*)(ws + OFF_FFMID);

  // Input projection + positional encoding.
  {
    const size_t nx = (size_t)MROWS * 64;
    f2bf_kernel<<<(unsigned)((nx + 255) / 256), 256, 0, stream>>>(x, xbf, nx);
    launch_gemm(0, xbf, w_in, b_in, h, nullptr, MROWS, DD, 64, 64, stream);
    const size_t nh = (size_t)MROWS * DD;
    pe_kernel<<<(unsigned)((nh + 255) / 256), 256, 0, stream>>>(h, hbf, pe, ps, pb);
  }

  for (int l = 0; l < LL; ++l) {
    const size_t wofs  = (size_t)l * DD * DD;
    const size_t bofs  = (size_t)l * DD;
    // QKV projections (bf16 outputs for the WMMA attention kernel).
    launch_gemm(0, hbf, wq + wofs, bq + bofs, nullptr, qbf, MROWS, DD, DD, DD, stream);
    launch_gemm(0, hbf, wk + wofs, bk + bofs, nullptr, kbf, MROWS, DD, DD, DD, stream);
    launch_gemm(0, hbf, wv + wofs, bv + bofs, nullptr, vbf, MROWS, DD, DD, DD, stream);
    // Fused attention.
    attention_kernel<<<dim3(SPAD / 64, BB * HH), 128, 0, stream>>>(
        qbf, kbf, vbf, tbias, obf);
    // Output projection (f32 for gated blend).
    launch_gemm(0, obf, wo + wofs, bo + bofs, attn, nullptr, MROWS, DD, DD, DD, stream);
    // Gate: relu(h@g1+gb1) @ g2 + gb2, sigmoid.
    launch_gemm(1, hbf, g1 + (size_t)l * DD * 256, gb1 + (size_t)l * 256,
                nullptr, ghid, MROWS, 256, DD, DD, stream);
    gate2_kernel<<<MROWS / 8, 256, 0, stream>>>(ghid, g2 + (size_t)l * 256,
                                                gb2 + l, gvec, MROWS);
    // Blend + double LN (in-place h / hbf update).
    ln_blend_kernel<<<MROWS, 128, 0, stream>>>(
        h, attn, gvec, lna_s + bofs, lna_b + bofs, n1s + bofs, n1b + bofs, hbf);
    // FFN with exact GELU.
    launch_gemm(2, hbf, f1 + (size_t)l * DD * DFF, fb1 + (size_t)l * DFF,
                nullptr, ffmid, MROWS, DFF, DD, DD, stream);
    launch_gemm(0, ffmid, f2 + (size_t)l * DFF * DD, fb2 + bofs,
                ff, nullptr, MROWS, DD, DFF, DFF, stream);
    ln_kernel<<<MROWS, 128, 0, stream>>>(h, ff, n2s + bofs, n2b + bofs, h, hbf);
  }

  // Final LayerNorm into d_out (f32).
  ln_kernel<<<MROWS, 128, 0, stream>>>(h, nullptr, fscale, fbias,
                                       (float*)d_out, nullptr);
}
